// CombinedHiddenEncoder_26800595927062
// MI455X (gfx1250) — compile-verified, matrix operand order checked by reading, not run on hardware
//
#include <hip/hip_runtime.h>
#include <cstdint>

// CDNA5 / gfx1250: wave32, WMMA 16x16x32 bf16 (f32 accumulate).
typedef __attribute__((ext_vector_type(16))) __bf16 v16bf;
typedef __attribute__((ext_vector_type(8)))  float  v8f;

// ---------------- degree / normalization ----------------

__global__ void deg_init_k(float* __restrict__ deg, int n) {
  int i = blockIdx.x * blockDim.x + threadIdx.x;
  if (i < n) deg[i] = 1.0f;                // self-loop contributes 1
}

__global__ void deg_accum_k(const int* __restrict__ dst, int e, float* __restrict__ deg) {
  int i = blockIdx.x * blockDim.x + threadIdx.x;
  if (i < e) atomicAdd(&deg[dst[i]], 1.0f);
}

__global__ void dinv_k(float* __restrict__ deg, int n) {
  int i = blockIdx.x * blockDim.x + threadIdx.x;
  if (i < n) deg[i] = rsqrtf(deg[i]);      // deg >= 1 always (self loop)
}

// ---------------- GEMM: C[:, col0:col0+16*colTiles] = A @ W ----------------
// One wave per 16x16 output tile. bf16 hi/lo split -> near-fp32 via 3 WMMAs.
// A: [rows, lda] fp32 row-major, K multiple of 32. W: [K, ldw] fp32 row-major.
__global__ __launch_bounds__(128)
void gemm_bf16x3_wmma(const float* __restrict__ A, int lda, int K,
                      const float* __restrict__ W, int ldw,
                      float* __restrict__ C, int ldc, int col0,
                      int rowTiles, int colTiles)
{
  int gwave = (blockIdx.x * blockDim.x + threadIdx.x) >> 5;
  int lane  = threadIdx.x & 31;
  if (gwave >= rowTiles * colTiles) return;          // whole-wave uniform exit
  int ct = gwave % colTiles;
  int rt = gwave / colTiles;
  int r  = rt * 16 + (lane & 15);                    // A row for this lane
  int cB = ct * 16 + (lane & 15);                    // B column for this lane
  int kb = (lane >> 4) << 3;                         // K sub-base: 0 or 8

  v8f acc = {};
  for (int k0 = 0; k0 < K; k0 += 32) {
    v16bf ah, al, bh, bl;
    const float* Ap = A + (size_t)r * lda + (k0 + kb);
    const float* Wp = W + (size_t)(k0 + kb) * ldw + cB;
#pragma unroll
    for (int j = 0; j < 8; ++j) {
      // A fragment (16-bit A 16x32 layout): elem j -> K=kb+j, elem 8+j -> K=kb+16+j
      float x0 = Ap[j];
      float x1 = Ap[16 + j];
      __bf16 h0 = (__bf16)x0, h1 = (__bf16)x1;
      ah[j]     = h0; al[j]     = (__bf16)(x0 - (float)h0);
      ah[8 + j] = h1; al[8 + j] = (__bf16)(x1 - (float)h1);
      // B fragment (transpose-symmetric: lane = column, same K pattern)
      float w0 = Wp[(size_t)j * ldw];
      float w1 = Wp[(size_t)(16 + j) * ldw];
      __bf16 g0 = (__bf16)w0, g1 = (__bf16)w1;
      bh[j]     = g0; bl[j]     = (__bf16)(w0 - (float)g0);
      bh[8 + j] = g1; bl[8 + j] = (__bf16)(w1 - (float)g1);
    }
    acc = __builtin_amdgcn_wmma_f32_16x16x32_bf16(false, ah, false, bh, (short)0, acc, false, false);
    acc = __builtin_amdgcn_wmma_f32_16x16x32_bf16(false, al, false, bh, (short)0, acc, false, false);
    acc = __builtin_amdgcn_wmma_f32_16x16x32_bf16(false, ah, false, bl, (short)0, acc, false, false);
  }
  // C/D layout: VGPR v -> row rt*16 + v + 8*(lane>=16), col = lane&15
  int rbase = rt * 16 + ((lane >> 4) << 3);
  int c = col0 + ct * 16 + (lane & 15);
#pragma unroll
  for (int v = 0; v < 8; ++v)
    C[(size_t)(rbase + v) * ldc + c] = acc[v];
}

// ---------------- propagation: init (self-loop + bias), edge scatter ----------------

__global__ void prop_init_k(const float* __restrict__ h, const float* __restrict__ dinv,
                            const float* __restrict__ bias0, const float* __restrict__ bias1,
                            int split, int D, float* __restrict__ out, int n)
{
  int idx = blockIdx.x * blockDim.x + threadIdx.x;
  if (idx >= n * D) return;
  int i = idx / D, c = idx % D;
  float w = dinv[i];
  float b = (c < split) ? bias0[c] : bias1[c - split];
  out[idx] = h[idx] * w * w + b;
}

__global__ __launch_bounds__(256)
void prop_edges_k(const int* __restrict__ src, const int* __restrict__ dst,
                  const float* __restrict__ dinv, const float* __restrict__ h,
                  float* __restrict__ out, int D, int E)
{
  int tid    = blockIdx.x * blockDim.x + threadIdx.x;
  int wave   = tid >> 5;
  int lane   = tid & 31;
  int nwaves = (gridDim.x * blockDim.x) >> 5;
  int chunk  = D >> 5;                           // floats per lane per row
  for (int e = wave; e < E; e += nwaves) {
    int s = src[e], d = dst[e];
    if (e + nwaves < E) {                        // hide random-gather latency
      int s2 = src[e + nwaves];
      __builtin_prefetch(h + (size_t)s2 * D + lane * chunk, 0, 1);
    }
    float nrm = dinv[s] * dinv[d];
    const float* hs = h   + (size_t)s * D;
    float*       od = out + (size_t)d * D;
    for (int c = lane; c < D; c += 32)
      atomicAdd(&od[c], hs[c] * nrm);            // L2-resident (out <= 51MB << 192MB L2)
  }
}

// ---------------- reparameterization + output pack ----------------

__global__ void finalize_k(const float* __restrict__ mv, const float* __restrict__ noise,
                           float* __restrict__ out, int n)
{
  int idx = blockIdx.x * blockDim.x + threadIdx.x;
  if (idx >= n * 64) return;
  int i = idx >> 6, c = idx & 63;
  float mean = mv[(size_t)i * 128 + c];
  float logv = mv[(size_t)i * 128 + 64 + c];
  out[idx]                       = noise[idx] * expf(0.5f * logv) + mean;  // z
  out[(size_t)n * 64 + idx]      = mean;
  out[(size_t)2 * n * 64 + idx]  = logv;
}

// ---------------- launch ----------------

extern "C" void kernel_launch(void* const* d_in, const int* in_sizes, int n_in,
                              void* d_out, int out_size, void* d_ws, size_t ws_size,
                              hipStream_t stream)
{
  (void)n_in; (void)out_size; (void)ws_size;
  const int FD = 256, CD = 128, HD = 128, LD = 64;

  const float* feature   = (const float*)d_in[0];
  const float* condition = (const float*)d_in[1];
  const int*   eidx      = (const int*)d_in[2];
  const float* noise     = (const float*)d_in[3];
  const float* W1 = (const float*)d_in[4];
  const float* b1 = (const float*)d_in[5];
  const float* W2 = (const float*)d_in[6];
  const float* b2 = (const float*)d_in[7];
  const float* W3 = (const float*)d_in[8];
  const float* b3 = (const float*)d_in[9];
  const float* Wm = (const float*)d_in[10];
  const float* bm = (const float*)d_in[11];
  const float* Wv = (const float*)d_in[12];
  const float* bv = (const float*)d_in[13];

  const int N = in_sizes[0] / FD;     // 50000 (divisible by 16 -> exact WMMA tiles)
  const int E = in_sizes[2] / 2;      // 600000
  const int* src = eidx;
  const int* dst = eidx + E;

  // workspace layout (all fp32):
  //   dinv [N] | buf0 [N x 256] | buf1 [N x 256] | buf2 [N x 128] | buf3 [N x 128]
  float* dinv = (float*)d_ws;
  float* buf0 = dinv + (((size_t)N + 511) & ~(size_t)511);
  float* buf1 = buf0 + (size_t)N * (2 * HD);
  float* buf2 = buf1 + (size_t)N * (2 * HD);
  float* buf3 = buf2 + (size_t)N * HD;

  auto cdiv = [](long long a, long long b) { return (int)((a + b - 1) / b); };

  // 1) symmetric normalization
  deg_init_k <<<cdiv(N, 256), 256, 0, stream>>>(dinv, N);
  deg_accum_k<<<cdiv(E, 256), 256, 0, stream>>>(dst, E, dinv);
  dinv_k     <<<cdiv(N, 256), 256, 0, stream>>>(dinv, N);

  const int rowTiles = N / 16;        // 3125

  // 2) layer 1 (fused f2h|c2h): buf0 = [X@W1 | C@W2]
  {
    int colTiles = HD / 16, waves = rowTiles * colTiles;
    gemm_bf16x3_wmma<<<cdiv((long long)waves * 32, 128), 128, 0, stream>>>(
        feature, FD, FD, W1, HD, buf0, 2 * HD, 0, rowTiles, colTiles);
    gemm_bf16x3_wmma<<<cdiv((long long)waves * 32, 128), 128, 0, stream>>>(
        condition, CD, CD, W2, HD, buf0, 2 * HD, HD, rowTiles, colTiles);
  }
  // propagate 256-wide: buf1 = A_norm @ buf0 + [b1|b2]
  prop_init_k <<<cdiv((long long)N * 2 * HD, 256), 256, 0, stream>>>(buf0, dinv, b1, b2, HD, 2 * HD, buf1, N);
  prop_edges_k<<<cdiv(E, 8), 256, 0, stream>>>(src, dst, dinv, buf0, buf1, 2 * HD, E);

  // 3) layer 2: buf2 = buf1 @ W3 ; buf3 = A_norm @ buf2 + b3
  {
    int colTiles = HD / 16, waves = rowTiles * colTiles;
    gemm_bf16x3_wmma<<<cdiv((long long)waves * 32, 128), 128, 0, stream>>>(
        buf1, 2 * HD, 2 * HD, W3, HD, buf2, HD, 0, rowTiles, colTiles);
  }
  prop_init_k <<<cdiv((long long)N * HD, 256), 256, 0, stream>>>(buf2, dinv, b3, b3, HD, HD, buf3, N);
  prop_edges_k<<<cdiv(E, 8), 256, 0, stream>>>(src, dst, dinv, buf2, buf3, HD, E);

  // 4) heads (fused mean|logvar): buf2 = [buf3@Wm | buf3@Wv] ; buf0 = A_norm @ buf2 + [bm|bv]
  {
    int colTiles = LD / 16, waves = rowTiles * colTiles;
    gemm_bf16x3_wmma<<<cdiv((long long)waves * 32, 128), 128, 0, stream>>>(
        buf3, HD, HD, Wm, LD, buf2, 2 * LD, 0, rowTiles, colTiles);
    gemm_bf16x3_wmma<<<cdiv((long long)waves * 32, 128), 128, 0, stream>>>(
        buf3, HD, HD, Wv, LD, buf2, 2 * LD, LD, rowTiles, colTiles);
  }
  prop_init_k <<<cdiv((long long)N * 2 * LD, 256), 256, 0, stream>>>(buf2, dinv, bm, bv, LD, 2 * LD, buf0, N);
  prop_edges_k<<<cdiv(E, 8), 256, 0, stream>>>(src, dst, dinv, buf2, buf0, 2 * LD, E);

  // 5) z = noise*exp(0.5*logvar) + mean ; pack (z, mean, logvar)
  finalize_k<<<cdiv((long long)N * LD, 256), 256, 0, stream>>>(buf0, noise, (float*)d_out, N);
}
